// Decoder_53317724013157
// MI455X (gfx1250) — matrix operand order repaired and checked by reading, not compiled, not run on hardware
//
#include <hip/hip_runtime.h>
#include <hip/hip_bf16.h>
#include <math.h>
#include <stdint.h>

// ---------------- problem constants (from reference) ----------------
#define VV   32000   // vocab
#define EE   512     // embedding dim
#define H2D  1024    // encoder hidden
#define DHD  1024    // decoder hidden
#define BB   64      // batch
#define SS   512     // src len
#define NAA  2       // num attention sources
#define LL   4       // gru layers
#define XK0  (NAA*H2D + EE)          // 2560, layer-0 GRU input dim
#define KOUT (NAA*H2D + EE + DHD)    // 3584, output projection K

typedef __attribute__((ext_vector_type(16))) _Float16 v16h;
typedef __attribute__((ext_vector_type(8)))  float    v8f;

// =====================================================================
// Elementwise / small kernels
// =====================================================================

__global__ void k_embed(const int* __restrict__ tok, const float* __restrict__ emb,
                        float* __restrict__ embedded) {
  int idx = blockIdx.x * blockDim.x + threadIdx.x;            // B*E
  if (idx >= BB * EE) return;
  int e = idx % EE, b = idx / EE;
  int t = tok[b];
  if (t >= VV) t = 0;                                         // UNK = 0
  embedded[idx] = emb[(size_t)t * EE + e];
}

__global__ void k_eaT(const float* __restrict__ ea, float* __restrict__ eaT) {
  int idx = blockIdx.x * blockDim.x + threadIdx.x;            // DH*B
  if (idx >= DHD * BB) return;
  int b = idx % BB, d = idx / BB;
  eaT[idx] = ea[(size_t)b * DHD + d];
}

// Pre-convert attn_W2 (rows DH..DH+H2) to f16: W2h[na][k][d]
__global__ void k_cvt_w2(const float* __restrict__ attn_W, _Float16* __restrict__ W2h) {
  int idx = blockIdx.x * blockDim.x + threadIdx.x;            // NA*H2*DH
  if (idx >= NAA * H2D * DHD) return;
  int rem = idx % (H2D * DHD);
  int na  = idx / (H2D * DHD);
  W2h[idx] = (_Float16)attn_W[(size_t)na * (DHD + H2D) * DHD + (size_t)DHD * DHD + rem];
}

// hW[na][b][d] = sum_k ea[b][k] * attn_W[na][k][d] + attn_b[na][d]   (rows 0..DH-1)
__global__ void k_hw(const float* __restrict__ ea, const float* __restrict__ attn_W,
                     const float* __restrict__ attn_b, float* __restrict__ hW) {
  int idx = blockIdx.x * blockDim.x + threadIdx.x;            // NA*B*DH
  if (idx >= NAA * BB * DHD) return;
  int d = idx % DHD;
  int b = (idx / DHD) % BB;
  int na = idx / (DHD * BB);
  const float* W1 = attn_W + (size_t)na * (DHD + H2D) * DHD;
  const float* h  = ea + (size_t)b * DHD;
  float acc = attn_b[na * DHD + d];
  for (int k = 0; k < DHD; ++k) acc += h[k] * W1[(size_t)k * DHD + d];
  hW[idx] = acc;
}

// xT0[k][b]: k<E -> embedded, else weighted
__global__ void k_x0T(const float* __restrict__ embedded, const float* __restrict__ weighted,
                      float* __restrict__ xT0) {
  int idx = blockIdx.x * blockDim.x + threadIdx.x;            // XK0*B
  if (idx >= XK0 * BB) return;
  int b = idx % BB, k = idx / BB;
  xT0[idx] = (k < EE) ? embedded[(size_t)b * EE + k]
                      : weighted[(size_t)b * (NAA * H2D) + (k - EE)];
}

// featT[k][b]: rows [0,DH)=h4T, [DH,DH+NA*H2)=weighted, rest=embedded
__global__ void k_featT(const float* __restrict__ hT, const float* __restrict__ weighted,
                        const float* __restrict__ embedded, float* __restrict__ featT) {
  int idx = blockIdx.x * blockDim.x + threadIdx.x;            // KOUT*B
  if (idx >= KOUT * BB) return;
  int b = idx % BB, k = idx / BB;
  float v;
  if (k < DHD)                 v = hT[(size_t)k * BB + b];
  else if (k < DHD + NAA*H2D)  v = weighted[(size_t)b * (NAA * H2D) + (k - DHD)];
  else                         v = embedded[(size_t)b * EE + (k - DHD - NAA * H2D)];
  featT[idx] = v;
}

// GRU gate math (torch order r,z,n); h = encoder_agenda broadcast
__global__ void k_gates(const float* __restrict__ giT, const float* __restrict__ ghT,
                        const float* __restrict__ b_ih, const float* __restrict__ b_hh,
                        const float* __restrict__ ea, float* __restrict__ hid_out,
                        float* __restrict__ hT, int l) {
  int idx = blockIdx.x * blockDim.x + threadIdx.x;            // B*DH
  if (idx >= BB * DHD) return;
  int d = idx % DHD, b = idx / DHD;
  const float* bi = b_ih + (size_t)l * 3 * DHD;
  const float* bh = b_hh + (size_t)l * 3 * DHD;
  float gir = giT[(size_t)d * BB + b]             + bi[d];
  float giz = giT[(size_t)(DHD + d) * BB + b]     + bi[DHD + d];
  float gin = giT[(size_t)(2 * DHD + d) * BB + b] + bi[2 * DHD + d];
  float ghr = ghT[(size_t)d * BB + b]             + bh[d];
  float ghz = ghT[(size_t)(DHD + d) * BB + b]     + bh[DHD + d];
  float ghn = ghT[(size_t)(2 * DHD + d) * BB + b] + bh[2 * DHD + d];
  float r = 1.f / (1.f + __expf(-(gir + ghr)));
  float z = 1.f / (1.f + __expf(-(giz + ghz)));
  float n = tanhf(gin + r * ghn);
  float h = ea[(size_t)b * DHD + d];
  float hn = (1.f - z) * n + z * h;
  hid_out[((size_t)l * BB + b) * DHD + d] = hn;
  hT[(size_t)d * BB + b] = hn;
}

// softmax over S per (na,b); reads masked scores from output buffer, writes probs
__global__ __launch_bounds__(256)
void k_softmax(const float* __restrict__ scores, float* __restrict__ probs) {
  int na = blockIdx.x / BB, b = blockIdx.x % BB;
  __shared__ float red[256];
  const float* srow = scores + ((size_t)na * BB + b) * SS;
  float*       prow = probs  + ((size_t)na * BB + b) * SS;
  int tid = threadIdx.x;
  float mx = -1e30f;
  for (int s = tid; s < SS; s += 256) mx = fmaxf(mx, srow[s]);
  red[tid] = mx; __syncthreads();
  for (int off = 128; off > 0; off >>= 1) {
    if (tid < off) red[tid] = fmaxf(red[tid], red[tid + off]);
    __syncthreads();
  }
  mx = red[0]; __syncthreads();
  float v0 = __expf(srow[tid] - mx);
  float v1 = __expf(srow[tid + 256] - mx);
  red[tid] = v0 + v1; __syncthreads();
  for (int off = 128; off > 0; off >>= 1) {
    if (tid < off) red[tid] += red[tid + off];
    __syncthreads();
  }
  float inv = 1.f / red[0];
  prow[tid]       = v0 * inv;
  prow[tid + 256] = v1 * inv;
}

// weighted[b][na*H2+d] = sum_s probs[na][b][s] * eo[na][s][b][d]
// grid: (NA*B, H2/256), 256 threads
__global__ __launch_bounds__(256)
void k_weighted(const float* __restrict__ probs, const float* __restrict__ eo,
                float* __restrict__ weighted) {
  int na = blockIdx.x / BB, b = blockIdx.x % BB;
  int d  = blockIdx.y * 256 + threadIdx.x;
  __shared__ float p[SS];
  for (int s = threadIdx.x; s < SS; s += 256)
    p[s] = probs[((size_t)na * BB + b) * SS + s];
  __syncthreads();
  const float* ep = eo + ((size_t)na * SS * BB + b) * H2D + d;   // + s*B*H2
  float acc = 0.f;
  for (int s = 0; s < SS; ++s) acc += p[s] * ep[(size_t)s * BB * H2D];
  weighted[(size_t)b * (NAA * H2D) + na * H2D + d] = acc;
}

// =====================================================================
// WMMA kernels
// =====================================================================

// Attention scores: energy = eo@W2h, fused epilogue score = sum_n v[n]*tanh(energy+hW)
// grid: (S*B/16, NA), 256 threads (8 waves); wave w covers N cols [w*128, w*128+128)
// A tile staged into LDS via CDNA5 async DMA (GLOBAL_LOAD_ASYNC_TO_LDS_B128).
__global__ __launch_bounds__(256)
void k_attn_scores(const float* __restrict__ eo, const _Float16* __restrict__ W2h,
                   const float* __restrict__ attn_v, const float* __restrict__ hW,
                   const float* __restrict__ mask, float* __restrict__ scores_out) {
  const int na = blockIdx.y;
  const int r0 = blockIdx.x * 16;                 // row block over S*B
  const int tid = threadIdx.x;
  const int wave = tid >> 5, lane = tid & 31;
  const int lane_lo = lane & 15, lane_hi = lane >> 4;
  __shared__ float Asf[16][32];                   // f32 A tile, filled by async DMA
  __shared__ float red[16];
  const float*    A  = eo  + (size_t)na * SS * BB * H2D;         // [S*B, H2] f32
  const _Float16* W2 = W2h + (size_t)na * H2D * DHD;             // [H2, DH]  f16

  // per-thread async source/dest for the A tile: threads 0..127, one B128 each
  const int am = (tid >> 3) & 15;                 // row 0..15
  const int ac = (tid & 7) * 4;                   // 4-float column chunk
  const unsigned lds_dst = (unsigned)(uintptr_t)(&Asf[am][ac]);  // LDS offset = addr[31:0]

  v8f acc[8] = {};
  for (int k0 = 0; k0 < H2D; k0 += 32) {
    __syncthreads();                              // previous tile fully consumed
    if (tid < 128) {
      const float* gp = &A[(size_t)(r0 + am) * H2D + k0 + ac];
      asm volatile("global_load_async_to_lds_b128 %0, %1, off"
                   :: "v"(lds_dst), "v"(gp) : "memory");
    }
    asm volatile("s_wait_asynccnt 0x0" ::: "memory");
    __syncthreads();                              // tile visible to all waves

    v16h a;                                       // convert during fragment build
#pragma unroll
    for (int j = 0; j < 8; ++j) {
      a[j]     = (_Float16)Asf[lane_lo][lane_hi * 8 + j];
      a[j + 8] = (_Float16)Asf[lane_lo][lane_hi * 8 + 16 + j];
    }
    // software-pipelined B fragments: load t+1 before WMMA on t
    const _Float16* wrow = W2 + (size_t)k0 * DHD + wave * 128 + lane_lo;
    v16h bf;
#pragma unroll
    for (int j = 0; j < 8; ++j) {
      bf[j]     = wrow[(size_t)(lane_hi * 8 + j) * DHD];
      bf[j + 8] = wrow[(size_t)(lane_hi * 8 + 16 + j) * DHD];
    }
#pragma unroll
    for (int t = 0; t < 8; ++t) {
      v16h bnext = bf;
      if (t < 7) {
#pragma unroll
        for (int j = 0; j < 8; ++j) {
          bnext[j]     = wrow[(size_t)(lane_hi * 8 + j) * DHD + (t + 1) * 16];
          bnext[j + 8] = wrow[(size_t)(lane_hi * 8 + 16 + j) * DHD + (t + 1) * 16];
        }
      }
      acc[t] = __builtin_amdgcn_wmma_f32_16x16x32_f16(false, a, false, bf,
                                                      (short)0, acc[t], false, false);
      bf = bnext;
    }
  }
  if (tid < 16) red[tid] = 0.f;
  __syncthreads();
  const float* hWb = hW + (size_t)na * BB * DHD;
  const float* av  = attn_v + (size_t)na * DHD;
#pragma unroll
  for (int v = 0; v < 8; ++v) {
    int m = v + lane_hi * 8;
    int b = (r0 + m) % BB;
    const float* hwr = hWb + (size_t)b * DHD;
    float s = 0.f;
#pragma unroll
    for (int t = 0; t < 8; ++t) {
      int n = wave * 128 + t * 16 + lane_lo;
      s += av[n] * tanhf(acc[t][v] + hwr[n]);
    }
    atomicAdd(&red[m], s);
  }
  __syncthreads();
  if (tid < 16) {
    int r = r0 + tid, sidx = r / BB, b = r % BB;
    float sc = red[tid];
    float mk = mask[((size_t)na * BB + b) * SS + sidx];
    scores_out[((size_t)na * BB + b) * SS + sidx] = (mk > 0.f) ? sc : -1e10f;
  }
}

// outT[M][64] = Wm[M][K] @ xT[K][64]; grid M/128, 256 threads (wave -> 16 rows, 4 n-tiles)
__global__ __launch_bounds__(256)
void k_gemm_WxT(const float* __restrict__ Wm, const float* __restrict__ xT,
                float* __restrict__ outT, int K) {
  const int tid = threadIdx.x, wave = tid >> 5, lane = tid & 31;
  const int lane_lo = lane & 15, lane_hi = lane >> 4;
  const int m0 = blockIdx.x * 128 + wave * 16;
  __shared__ _Float16 Bs[32][64];
  v8f acc[4] = {};
  for (int k0 = 0; k0 < K; k0 += 32) {
    __syncthreads();
    for (int i = tid; i < 32 * 64; i += 256) {
      int kk = i >> 6, n = i & 63;
      Bs[kk][n] = (_Float16)xT[(size_t)(k0 + kk) * 64 + n];
    }
    __syncthreads();
    v16h a;
    const float* ap = Wm + (size_t)(m0 + lane_lo) * K + k0 + lane_hi * 8;
#pragma unroll
    for (int j = 0; j < 8; ++j) { a[j] = (_Float16)ap[j]; a[j + 8] = (_Float16)ap[16 + j]; }
#pragma unroll
    for (int t = 0; t < 4; ++t) {
      v16h bf;
#pragma unroll
      for (int j = 0; j < 8; ++j) { bf[j] = Bs[lane_hi * 8 + j][t * 16 + lane_lo]; bf[j + 8] = Bs[lane_hi * 8 + 16 + j][t * 16 + lane_lo]; }
      acc[t] = __builtin_amdgcn_wmma_f32_16x16x32_f16(false, a, false, bf,
                                                      (short)0, acc[t], false, false);
    }
  }
#pragma unroll
  for (int t = 0; t < 4; ++t)
#pragma unroll
    for (int v = 0; v < 8; ++v) {
      int m = m0 + v + lane_hi * 8;
      int n = t * 16 + lane_lo;
      outT[(size_t)m * 64 + n] = acc[t][v];
    }
}

// logits[b][v] = out_W[v][:] . feat[b][:] + out_b[v]; computed as out_W @ featT
__global__ __launch_bounds__(256)
void k_out_proj(const float* __restrict__ outW, const float* __restrict__ featT,
                const float* __restrict__ outb, float* __restrict__ logits) {
  const int tid = threadIdx.x, wave = tid >> 5, lane = tid & 31;
  const int lane_lo = lane & 15, lane_hi = lane >> 4;
  const int m0 = blockIdx.x * 128 + wave * 16;                  // vocab rows
  __shared__ _Float16 Bs[32][64];
  v8f acc[4] = {};
  for (int k0 = 0; k0 < KOUT; k0 += 32) {
    __syncthreads();
    for (int i = tid; i < 32 * 64; i += 256) {
      int kk = i >> 6, n = i & 63;
      Bs[kk][n] = (_Float16)featT[(size_t)(k0 + kk) * 64 + n];
    }
    __syncthreads();
    v16h a;
    const float* ap = outW + (size_t)(m0 + lane_lo) * KOUT + k0 + lane_hi * 8;
#pragma unroll
    for (int j = 0; j < 8; ++j) { a[j] = (_Float16)ap[j]; a[j + 8] = (_Float16)ap[16 + j]; }
#pragma unroll
    for (int t = 0; t < 4; ++t) {
      v16h bf;
#pragma unroll
      for (int j = 0; j < 8; ++j) { bf[j] = Bs[lane_hi * 8 + j][t * 16 + lane_lo]; bf[j + 8] = Bs[lane_hi * 8 + 16 + j][t * 16 + lane_lo]; }
      acc[t] = __builtin_amdgcn_wmma_f32_16x16x32_f16(false, a, false, bf,
                                                      (short)0, acc[t], false, false);
    }
  }
#pragma unroll
  for (int t = 0; t < 4; ++t)
#pragma unroll
    for (int v = 0; v < 8; ++v) {
      int m = m0 + v + lane_hi * 8;                              // vocab index
      int b = t * 16 + lane_lo;                                  // batch index
      logits[(size_t)b * VV + m] = acc[t][v] + outb[m];
    }
}

// =====================================================================
// Launch
// =====================================================================
extern "C" void kernel_launch(void* const* d_in, const int* in_sizes, int n_in,
                              void* d_out, int out_size, void* d_ws, size_t ws_size,
                              hipStream_t stream) {
  (void)in_sizes; (void)n_in; (void)out_size; (void)ws_size;
  const int*   tok     = (const int*)  d_in[0];
  const float* ea      = (const float*)d_in[1];
  const float* eo      = (const float*)d_in[2];
  const float* mask    = (const float*)d_in[3];
  const float* emb     = (const float*)d_in[4];
  const float* attn_W  = (const float*)d_in[5];
  const float* attn_b  = (const float*)d_in[6];
  const float* attn_v  = (const float*)d_in[7];
  const float* W_ih0   = (const float*)d_in[8];
  const float* W_ihr   = (const float*)d_in[9];
  const float* W_hh    = (const float*)d_in[10];
  const float* b_ih    = (const float*)d_in[11];
  const float* b_hh    = (const float*)d_in[12];
  const float* out_W   = (const float*)d_in[13];
  const float* out_b   = (const float*)d_in[14];

  float* logits_out = (float*)d_out;
  float* hidden_out = logits_out + (size_t)BB * VV;
  float* scores_out = hidden_out + (size_t)LL * BB * DHD;

  float* ws = (float*)d_ws;
  float* hW       = ws;                 size_t o = (size_t)NAA * BB * DHD;   // 131072
  float* embedded = ws + o;             o += (size_t)BB * EE;                // +32768
  float* weighted = ws + o;             o += (size_t)BB * NAA * H2D;         // +131072
  float* xT0      = ws + o;             o += (size_t)XK0 * BB;               // +163840
  float* eaT      = ws + o;             o += (size_t)DHD * BB;               // +65536
  float* hT       = ws + o;             o += (size_t)DHD * BB;               // +65536
  float* giT      = ws + o;             o += (size_t)3 * DHD * BB;           // +196608
  float* ghT      = ws + o;             o += (size_t)3 * DHD * BB;           // +196608
  float* featT    = ws + o;             o += (size_t)KOUT * BB;              // +229376
  float* probs    = ws + o;             o += (size_t)NAA * BB * SS;          // +65536
  _Float16* W2h   = (_Float16*)(ws + o);            // NA*H2*DH halfs (4 MB), 32B aligned

  k_cvt_w2<<<(NAA * H2D * DHD + 255) / 256, 256, 0, stream>>>(attn_W, W2h);
  k_embed<<<(BB * EE + 255) / 256, 256, 0, stream>>>(tok, emb, embedded);
  k_eaT  <<<(DHD * BB + 255) / 256, 256, 0, stream>>>(ea, eaT);
  k_hw   <<<(NAA * BB * DHD + 255) / 256, 256, 0, stream>>>(ea, attn_W, attn_b, hW);

  k_attn_scores<<<dim3((SS * BB) / 16, NAA), 256, 0, stream>>>(eo, W2h, attn_v, hW,
                                                               mask, scores_out);
  k_softmax<<<NAA * BB, 256, 0, stream>>>(scores_out, probs);
  k_weighted<<<dim3(NAA * BB, H2D / 256), 256, 0, stream>>>(probs, eo, weighted);
  k_x0T<<<(XK0 * BB + 255) / 256, 256, 0, stream>>>(embedded, weighted, xT0);

  for (int l = 0; l < LL; ++l) {
    const float* Wi = (l == 0) ? W_ih0 : (W_ihr + (size_t)(l - 1) * 3 * DHD * DHD);
    const float* xt = (l == 0) ? xT0 : hT;
    int K = (l == 0) ? XK0 : DHD;
    k_gemm_WxT<<<(3 * DHD) / 128, 256, 0, stream>>>(Wi, xt, giT, K);
    k_gemm_WxT<<<(3 * DHD) / 128, 256, 0, stream>>>(W_hh + (size_t)l * 3 * DHD * DHD,
                                                    eaT, ghT, DHD);
    k_gates<<<(BB * DHD + 255) / 256, 256, 0, stream>>>(giT, ghT, b_ih, b_hh, ea,
                                                        hidden_out, hT, l);
  }

  k_featT<<<(KOUT * BB + 255) / 256, 256, 0, stream>>>(hT, weighted, embedded, featT);
  k_out_proj<<<VV / 128, 256, 0, stream>>>(out_W, featT, out_b, logits_out);
}
